// SingleAttention_4398046511646
// MI455X (gfx1250) — matrix operand order repaired and checked by reading, not compiled, not run on hardware
//
#include <hip/hip_runtime.h>
#include <math.h>

typedef __attribute__((ext_vector_type(16))) _Float16     v16h;
typedef __attribute__((ext_vector_type(8)))  _Float16     v8h;
typedef __attribute__((ext_vector_type(8)))  float        v8f;
typedef __attribute__((ext_vector_type(4)))  unsigned int u32x4;
typedef __attribute__((ext_vector_type(8)))  int          i32x8;
typedef __attribute__((ext_vector_type(4)))  int          i32x4;

#define B_  8
#define S_  2048
#define D_  512
#define DK_ 64

// fold 1/sqrt(DK) * log2(e) into q so attention uses exp2 (v_exp_f32)
#define QSCALE 0.18033688011112042f

union HFrag { v16h v; v8h h[2]; };

// ---------------------------------------------------------------------------
// TDM: issue a 2D f16 tile load (Global -> LDS) via the Tensor Data Mover.
// Descriptor layout per CDNA5 ISA §8.3/§8.4. Groups 2/3 zero (<=2D tensor).
//   tensor: rows of tdim0 elements (stride0 elements apart), tdim1 rows
//   tile:   tiled0 x tiled1 elements, written row-major to LDS at lds_addr
// ---------------------------------------------------------------------------
__device__ inline void tdm_load_2d_f16(unsigned lds_addr, const void* gptr,
                                       unsigned tdim0, unsigned tdim1,
                                       unsigned tiled0, unsigned tiled1,
                                       unsigned stride0)
{
    unsigned long long ga = (unsigned long long)(uintptr_t)gptr;
    u32x4 g0;
    g0[0] = 1u;                                            // count=1, user D#
    g0[1] = lds_addr;                                      // lds_addr[31:0]
    g0[2] = (unsigned)(ga & 0xffffffffu);                  // global_addr lo
    g0[3] = (unsigned)((ga >> 32) & 0x01ffffffu)           // global_addr hi
          | (2u << 30);                                    // type = 2 (image)
    i32x8 g1;
    g1[0] = (int)(1u << 16);                               // data_size=1 (2B)
    g1[1] = (int)((tdim0 & 0xffffu) << 16);                // tensor_dim0 lo16
    g1[2] = (int)((tdim0 >> 16) | ((tdim1 & 0xffffu) << 16));
    g1[3] = (int)((tdim1 >> 16) | (tiled0 << 16));         // tile_dim0
    g1[4] = (int)(tiled1 & 0xffffu);                       // tile_dim1 (dim2=0)
    g1[5] = (int)stride0;                                  // dim0_stride lo32
    g1[6] = 0;                                             // stride hi, dim1_stride
    g1[7] = 0;
    i32x4 z4 = {0, 0, 0, 0};
#if __clang_major__ >= 23
    i32x8 z8 = {0, 0, 0, 0, 0, 0, 0, 0};
    __builtin_amdgcn_tensor_load_to_lds(g0, g1, z4, z4, z8, 0);
#else
    __builtin_amdgcn_tensor_load_to_lds(g0, g1, z4, z4, 0);
#endif
}

// ---------------------------------------------------------------------------
// Kernel 1: repack W[512][64] fp32 -> f16 in WMMA B-fragment order:
//   wpack[which][(kb*64 + n)*32 + kk] = (f16) W[kb*32+kk][n]
// ---------------------------------------------------------------------------
__global__ __launch_bounds__(256) void prepack_w_kernel(
    const float* __restrict__ Wq, const float* __restrict__ Wk,
    const float* __restrict__ Wv, _Float16* __restrict__ wpack)
{
    const int which = blockIdx.y;
    const float* W = (which == 0) ? Wq : (which == 1) ? Wk : Wv;
    int idx = blockIdx.x * 256 + threadIdx.x;      // 0 .. 32767
    int kk = idx & 31;
    int n  = (idx >> 5) & 63;
    int kb = idx >> 11;                            // 0 .. 15
    wpack[(size_t)which * (D_ * DK_) + ((size_t)(kb * 64 + n) * 32) + kk] =
        (_Float16)W[(size_t)(kb * 32 + kk) * DK_ + n];
}

// ---------------------------------------------------------------------------
// Kernel 2: projections. One wave computes a 16-row x 64-col tile of
//   y = x @ W + b  using v_wmma_f32_16x16x32_f16, K-loop over D=512.
// which==0: q -> f16 [B*S][64], scaled by QSCALE
// which==1: k -> f16 [B*S][64]
// which==2: v -> f16 transposed vt[b][dv][s]
// ---------------------------------------------------------------------------
__global__ __launch_bounds__(256) void proj_kernel(
    const float* __restrict__ xq, const float* __restrict__ xk,
    const float* __restrict__ xv, const _Float16* __restrict__ wpack,
    const float* __restrict__ bq, const float* __restrict__ bk,
    const float* __restrict__ bv,
    _Float16* __restrict__ q16, _Float16* __restrict__ k16,
    _Float16* __restrict__ vt16)
{
    const int which = blockIdx.y;
    const float*    x    = (which == 0) ? xq : (which == 1) ? xk : xv;
    const _Float16* wp   = wpack + (size_t)which * (D_ * DK_);
    const float*    bias = (which == 0) ? bq : (which == 1) ? bk : bv;

    const int lane = threadIdx.x & 31;
    const int wave = threadIdx.x >> 5;
    const int tile = blockIdx.x * 8 + wave;        // 0 .. 1023
    const int row0 = tile * 16;                    // global row in [0, B*S)
    const int lh   = lane >> 4;                    // half-wave select
    const int ln   = lane & 15;

    v8f acc0 = {}, acc1 = {}, acc2 = {}, acc3 = {};

    const float* xrow = x + (size_t)(row0 + ln) * D_;

    for (int kb = 0; kb < D_ / 32; ++kb) {
        // A fragment: row = ln, K = lh*16 + 0..15 (contiguous, fp32 -> f16)
        HFrag a;
        const float* s0 = xrow + kb * 32 + lh * 16;
        const float* s1 = s0 + 8;
        #pragma unroll
        for (int i = 0; i < 8; ++i) {
            a.h[0][i] = (_Float16)s0[i];
            a.h[1][i] = (_Float16)s1[i];
        }
        // B fragments: col = c*16+ln, K = lh*16 + 0..15 (contiguous, prepacked)
        const _Float16* bp = wp + ((size_t)kb * 64 + ln) * 32 + lh * 16;
        v16h b0 = *(const v16h*)(bp);
        v16h b1 = *(const v16h*)(bp + 16 * 32);
        v16h b2 = *(const v16h*)(bp + 32 * 32);
        v16h b3 = *(const v16h*)(bp + 48 * 32);
        acc0 = __builtin_amdgcn_wmma_f32_16x16x32_f16(false, a.v, false, b0, (short)0, acc0, false, false);
        acc1 = __builtin_amdgcn_wmma_f32_16x16x32_f16(false, a.v, false, b1, (short)0, acc1, false, false);
        acc2 = __builtin_amdgcn_wmma_f32_16x16x32_f16(false, a.v, false, b2, (short)0, acc2, false, false);
        acc3 = __builtin_amdgcn_wmma_f32_16x16x32_f16(false, a.v, false, b3, (short)0, acc3, false, false);
    }

    // Epilogue: bias add (+scale for q), store f16
    #pragma unroll
    for (int c = 0; c < 4; ++c) {
        const v8f acc = (c == 0) ? acc0 : (c == 1) ? acc1 : (c == 2) ? acc2 : acc3;
        const int n = c * 16 + ln;
        const float bb = bias[n];
        #pragma unroll
        for (int e = 0; e < 8; ++e) {
            const int row = row0 + lh * 8 + e;     // C layout: M = lh*8 + e
            float val = acc[e] + bb;
            if (which == 0) {
                q16[(size_t)row * DK_ + n] = (_Float16)(val * QSCALE);
            } else if (which == 1) {
                k16[(size_t)row * DK_ + n] = (_Float16)val;
            } else {
                const int b = row >> 11;           // row / S_
                const int s = row & (S_ - 1);
                vt16[((size_t)b * DK_ + n) * S_ + s] = (_Float16)val;
            }
        }
    }
}

// ---------------------------------------------------------------------------
// Kernel 3: flash attention. 4 waves/block process 4 query tiles of the SAME
// batch; K/V tiles are staged into LDS once per block by the Tensor Data
// Mover (double-buffered, wave 0 issues, s_wait_tensorcnt + barriers sync),
// then each wave runs 4 WMMAs for scores, online softmax (exp2 domain), LDS
// lane-transpose of P, and 4 WMMAs for P@V.
// ---------------------------------------------------------------------------
__global__ __launch_bounds__(128) void attn_kernel(
    const _Float16* __restrict__ q16, const _Float16* __restrict__ k16,
    const _Float16* __restrict__ vt16, float* __restrict__ out)
{
    __shared__ __align__(32) _Float16 kbuf[2][32 * 64];    // K tile  [key][dk]
    __shared__ __align__(32) _Float16 vbuf[2][64 * 32];    // V^T tile [dv][key]
    __shared__ __align__(32) _Float16 lds_p[4][16 * 32];   // P transpose / wave

    const int lane = threadIdx.x & 31;
    const int wave = threadIdx.x >> 5;
    const int tile = blockIdx.x * 4 + wave;        // 0 .. 1023
    const int b    = tile >> 7;                    // 128 query-tiles / batch
    const int q0   = (tile & 127) * 16;
    const int lh   = lane >> 4;
    const int ln   = lane & 15;

    // Q A-fragments for both K-chunks of DK=64 (q already scaled by QSCALE)
    HFrag aq0, aq1;
    const _Float16* qrow = q16 + ((size_t)b * S_ + q0 + ln) * DK_;
    aq0.h[0] = *(const v8h*)(qrow + lh * 16);
    aq0.h[1] = *(const v8h*)(qrow + 8 + lh * 16);
    aq1.h[0] = *(const v8h*)(qrow + 32 + lh * 16);
    aq1.h[1] = *(const v8h*)(qrow + 40 + lh * 16);

    v8f o0 = {}, o1 = {}, o2 = {}, o3 = {};
    v8f mrow, lrow;
    #pragma unroll
    for (int e = 0; e < 8; ++e) { mrow[e] = -1e30f; lrow[e] = 0.0f; }

    const _Float16* kg = k16  + (size_t)b * S_ * DK_;      // [S][64] f16
    const _Float16* vg = vt16 + (size_t)b * DK_ * S_;      // [64][S] f16
    _Float16* plds = &lds_p[wave][0];

    // wave 0 issues both TDM descriptors for key-block starting at s0 -> buf
    auto issue_tiles = [&](int s0, int buf) {
        tdm_load_2d_f16((unsigned)(uintptr_t)&kbuf[buf][0],
                        kg + (size_t)s0 * DK_,
                        /*tdim0=*/DK_, /*tdim1=*/S_,
                        /*tile =*/DK_, 32, /*stride0=*/DK_);
        tdm_load_2d_f16((unsigned)(uintptr_t)&vbuf[buf][0],
                        vg + s0,
                        /*tdim0=*/S_, /*tdim1=*/DK_,
                        /*tile =*/32, DK_, /*stride0=*/S_);
    };

    if (wave == 0) issue_tiles(0, 0);

    for (int it = 0; it < S_ / 32; ++it) {
        const int cur = it & 1;
        if (wave == 0) __builtin_amdgcn_s_wait_tensorcnt(0);   // buf[cur] ready
        __syncthreads();
        if (wave == 0 && it + 1 < S_ / 32) issue_tiles((it + 1) * 32, cur ^ 1);

        const _Float16* kl = &kbuf[cur][0];
        const _Float16* vl = &vbuf[cur][0];

        // ---- scores: S[16x32] = Q[16x64] @ K^T (B-fragments from LDS) ----
        v8f sc0 = {}, sc1 = {};
        {
            v16h bk0 = *(const v16h*)(kl + (size_t)ln * DK_ + lh * 16);
            v16h bk1 = *(const v16h*)(kl + (size_t)ln * DK_ + 32 + lh * 16);
            sc0 = __builtin_amdgcn_wmma_f32_16x16x32_f16(false, aq0.v, false, bk0, (short)0, sc0, false, false);
            sc0 = __builtin_amdgcn_wmma_f32_16x16x32_f16(false, aq1.v, false, bk1, (short)0, sc0, false, false);
            v16h bk2 = *(const v16h*)(kl + (size_t)(16 + ln) * DK_ + lh * 16);
            v16h bk3 = *(const v16h*)(kl + (size_t)(16 + ln) * DK_ + 32 + lh * 16);
            sc1 = __builtin_amdgcn_wmma_f32_16x16x32_f16(false, aq0.v, false, bk2, (short)0, sc1, false, false);
            sc1 = __builtin_amdgcn_wmma_f32_16x16x32_f16(false, aq1.v, false, bk3, (short)0, sc1, false, false);
        }

        // ---- online softmax; butterfly over 16-lane halves matches C layout --
        v8f fac;
        #pragma unroll
        for (int e = 0; e < 8; ++e) {
            float t = fmaxf(sc0[e], sc1[e]);
            t = fmaxf(t, __shfl_xor(t, 1, 32));
            t = fmaxf(t, __shfl_xor(t, 2, 32));
            t = fmaxf(t, __shfl_xor(t, 4, 32));
            t = fmaxf(t, __shfl_xor(t, 8, 32));
            const float mn = fmaxf(mrow[e], t);
            const float p0 = exp2f(sc0[e] - mn);
            const float p1 = exp2f(sc1[e] - mn);
            float ps = p0 + p1;
            ps += __shfl_xor(ps, 1, 32);
            ps += __shfl_xor(ps, 2, 32);
            ps += __shfl_xor(ps, 4, 32);
            ps += __shfl_xor(ps, 8, 32);
            const float f = exp2f(mrow[e] - mn);
            lrow[e] = lrow[e] * f + ps;
            mrow[e] = mn;
            fac[e]  = f;
            // transpose P through LDS (f16), row M = lh*8+e, col = key idx
            const int M = lh * 8 + e;
            plds[M * 32 + ln]      = (_Float16)p0;
            plds[M * 32 + 16 + ln] = (_Float16)p1;
        }
        #pragma unroll
        for (int e = 0; e < 8; ++e) {
            o0[e] *= fac[e]; o1[e] *= fac[e]; o2[e] *= fac[e]; o3[e] *= fac[e];
        }

        // ---- reload P as A fragment (wave-local LDS; dscnt handled) ----
        HFrag pa;
        pa.h[0] = *(const v8h*)(plds + ln * 32 + lh * 16);
        pa.h[1] = *(const v8h*)(plds + ln * 32 + 8 + lh * 16);

        // ---- O += P[16x32] @ V[32x64]; V^T B-fragments from LDS ----
        v16h vb0 = *(const v16h*)(vl + (size_t)(0 * 16 + ln) * 32 + lh * 16);
        v16h vb1 = *(const v16h*)(vl + (size_t)(1 * 16 + ln) * 32 + lh * 16);
        v16h vb2 = *(const v16h*)(vl + (size_t)(2 * 16 + ln) * 32 + lh * 16);
        v16h vb3 = *(const v16h*)(vl + (size_t)(3 * 16 + ln) * 32 + lh * 16);
        o0 = __builtin_amdgcn_wmma_f32_16x16x32_f16(false, pa.v, false, vb0, (short)0, o0, false, false);
        o1 = __builtin_amdgcn_wmma_f32_16x16x32_f16(false, pa.v, false, vb1, (short)0, o1, false, false);
        o2 = __builtin_amdgcn_wmma_f32_16x16x32_f16(false, pa.v, false, vb2, (short)0, o2, false, false);
        o3 = __builtin_amdgcn_wmma_f32_16x16x32_f16(false, pa.v, false, vb3, (short)0, o3, false, false);

        __syncthreads();   // all waves done with buf[cur] before next overwrite
    }

    // ---- finalize: out[row][dv] = O / l ----
    float* orow = out + ((size_t)b * S_ + q0) * DK_;
    #pragma unroll
    for (int e = 0; e < 8; ++e) {
        const int M = lh * 8 + e;
        const float inv = 1.0f / lrow[e];
        orow[(size_t)M * DK_ + ln]      = o0[e] * inv;
        orow[(size_t)M * DK_ + 16 + ln] = o1[e] * inv;
        orow[(size_t)M * DK_ + 32 + ln] = o2[e] * inv;
        orow[(size_t)M * DK_ + 48 + ln] = o3[e] * inv;
    }
}

// ---------------------------------------------------------------------------
extern "C" void kernel_launch(void* const* d_in, const int* in_sizes, int n_in,
                              void* d_out, int out_size, void* d_ws, size_t ws_size,
                              hipStream_t stream)
{
    (void)in_sizes; (void)n_in; (void)out_size; (void)ws_size;

    const float* xq = (const float*)d_in[0];
    const float* xk = (const float*)d_in[1];
    const float* xv = (const float*)d_in[2];
    const float* Wq = (const float*)d_in[3];
    const float* bq = (const float*)d_in[4];
    const float* Wk = (const float*)d_in[5];
    const float* bk = (const float*)d_in[6];
    const float* Wv = (const float*)d_in[7];
    const float* bv = (const float*)d_in[8];
    float* out = (float*)d_out;

    char* ws = (char*)d_ws;
    _Float16* wpack = (_Float16*)(ws);                                 // 192 KB
    _Float16* q16   = (_Float16*)(ws + (256u << 10));                  // 2 MB
    _Float16* k16   = (_Float16*)(ws + (256u << 10) + (2u << 20));     // 2 MB
    _Float16* vt16  = (_Float16*)(ws + (256u << 10) + (4u << 20));     // 2 MB

    // 1) prepack weights into WMMA B-fragment order (f16)
    prepack_w_kernel<<<dim3(128, 3), 256, 0, stream>>>(Wq, Wk, Wv, wpack);

    // 2) projections: 1024 row-tiles per matrix, 8 waves/block
    proj_kernel<<<dim3(128, 3), 256, 0, stream>>>(
        xq, xk, xv, wpack, bq, bk, bv, q16, k16, vt16);

    // 3) flash attention: 1024 query tiles, 4 waves/block, TDM-staged K/V
    attn_kernel<<<dim3(256), 128, 0, stream>>>(q16, k16, vt16, out);
}
